// AttDecoder_68771016344085
// MI455X (gfx1250) — matrix operand order repaired and checked by reading, not compiled, not run on hardware
//
#include <hip/hip_runtime.h>
#include <hip/hip_bf16.h>
#include <math.h>

typedef __attribute__((ext_vector_type(16))) __bf16 v16bf;
typedef __attribute__((ext_vector_type(8)))  __bf16 v8bf;
typedef __attribute__((ext_vector_type(8)))  float  v8f;

constexpr int kB  = 32;
constexpr int kH  = 16;
constexpr int kW  = 64;
constexpr int kHW = 1024;          // H*W
constexpr int kP  = kB * kHW;      // 32768 positions
constexpr int kD  = 684;
constexpr int kDP = 704;           // D padded to multiple of 32
constexpr int kA  = 512;
constexpr int kHD = 256;
constexpr int kE  = 256;
constexpr int kV  = 111;
constexpr int kT  = 48;
constexpr int kMT = kP / 16;       // 2048 M-tiles

// ---------------- precompute: avg over spatial ----------------
__global__ void k_avg(const float* __restrict__ feat, float* __restrict__ avg) {
  int b = blockIdx.x;
  for (int d = threadIdx.x; d < kD; d += blockDim.x) {
    const float* f = feat + ((size_t)b * kD + d) * kHW;
    float s = 0.f;
    for (int i = 0; i < kHW; ++i) s += f[i];
    avg[b * kDP + d] = s * (1.0f / kHW);
  }
}

// ---------------- precompute: hidden0 = tanh(avg @ init_w^T + b) ----------------
__global__ void k_hidden0(const float* __restrict__ avg, const float* __restrict__ init_w,
                          const float* __restrict__ init_b, float* __restrict__ hidden) {
  int b = blockIdx.x;
  int h = threadIdx.x;                // 256 threads
  const float* w = init_w + (size_t)h * kD;
  const float* a = avg + b * kDP;
  float s = init_b[h];
  for (int d = 0; d < kD; ++d) s += w[d] * a[d];
  hidden[b * kHD + h] = tanhf(s);
}

// ---------------- precompute: features -> bf16 (p, K) row-major, zero-padded ----------------
__global__ void k_featT(const float* __restrict__ feat, __bf16* __restrict__ featT) {
  __shared__ float tile[16][17];
  int pt = blockIdx.x, dt = blockIdx.y;
  int tx = threadIdx.x, ty = threadIdx.y;
  int p0 = pt * 16, d0 = dt * 16;
  int p = p0 + tx;                     // hw fastest -> coalesced read
  int b = p >> 10, hw = p & 1023;
  int d = d0 + ty;
  float v = (d < kD) ? feat[((size_t)b * kD + d) * kHW + hw] : 0.f;
  tile[ty][tx] = v;
  __syncthreads();
  featT[(size_t)(p0 + ty) * kDP + d0 + tx] = (__bf16)tile[tx][ty];
}

// ---------------- precompute: Ua_w -> bf16 (a, K) row-major, zero-padded ----------------
__global__ void k_uawcvt(const float* __restrict__ ua_w, __bf16* __restrict__ uawb) {
  int a = blockIdx.x;
  for (int d = threadIdx.x; d < kDP; d += blockDim.x)
    uawb[(size_t)a * kDP + d] = (d < kD) ? (__bf16)ua_w[(size_t)a * kD + d] : (__bf16)0.f;
}

// ---------------- precompute: fused coverage kernel K[a, tap] = sum_c Uf[a,c]*convQ[c,tap] ----------------
__global__ void k_fusecover(const float* __restrict__ uf_w, const float* __restrict__ convq,
                            __bf16* __restrict__ kf) {
  int a = blockIdx.x;                 // 512
  int p = threadIdx.x;                // 128 (121 taps + pad)
  float s = 0.f;
  if (p < 121) {
    const float* u = uf_w + (size_t)a * kA;
    for (int c = 0; c < kA; ++c) s += u[c] * convq[c * 121 + p];
  }
  kf[a * 128 + p] = (__bf16)s;
}

// ---------------- precompute: Ua_feaT[a][p] = (featT @ uawb^T + Ua_b), bf16 ----------------
// One wave per block computes a 16x16 tile; K = 704 (22 chunks of 32) bf16 WMMA.
// All fragment feeds are contiguous b128 loads.
__global__ void k_uafea(const __bf16* __restrict__ featT, const __bf16* __restrict__ uawb,
                        const float* __restrict__ ua_b, __bf16* __restrict__ ua_feaT) {
  int mt = blockIdx.x;                // 2048
  int nt = blockIdx.y;                // 32
  int lane = threadIdx.x & 31;
  int hi = lane >> 4, ln = lane & 15;
  int p = mt * 16 + ln;
  int aCol = nt * 16 + ln;
  const __bf16* arow = featT + (size_t)p * kDP;
  const __bf16* brow = uawb + (size_t)aCol * kDP;
  v8f acc = {};
  for (int kc = 0; kc < 22; ++kc) {
    v8bf a0 = *(const v8bf*)(arow + kc * 32 + hi * 8);        // K [hi*8, hi*8+8)
    v8bf a1 = *(const v8bf*)(arow + kc * 32 + 16 + hi * 8);   // K [16+hi*8, ...)
    v16bf af = __builtin_shufflevector(a0, a1, 0, 1, 2, 3, 4, 5, 6, 7,
                                       8, 9, 10, 11, 12, 13, 14, 15);
    v8bf b0 = *(const v8bf*)(brow + kc * 32 + hi * 8);        // K [hi*8, hi*8+8)
    v8bf b1 = *(const v8bf*)(brow + kc * 32 + 16 + hi * 8);   // K [16+hi*8, ...)
    v16bf bfm = __builtin_shufflevector(b0, b1, 0, 1, 2, 3, 4, 5, 6, 7,
                                        8, 9, 10, 11, 12, 13, 14, 15);
    acc = __builtin_amdgcn_wmma_f32_16x16x32_bf16(false, af, false, bfm, (short)0, acc,
                                                  false, false);
  }
  __builtin_amdgcn_s_wait_tensorcnt(0);
  float bias = ua_b[aCol];
  v8bf outv;
#pragma unroll
  for (int v = 0; v < 8; ++v) outv[v] = (__bf16)(acc[v] + bias);
  // acc[v] is row m = hi*8 + v -> positions are contiguous: one b128 store
  *(v8bf*)(ua_feaT + (size_t)aCol * kP + mt * 16 + hi * 8) = outv;
}

// ---------------- per step: Wa_h = hidden @ Wa_w^T + Wa_b ----------------
__global__ void k_wah(const float* __restrict__ hidden, const float* __restrict__ wa_w,
                      const float* __restrict__ wa_b, float* __restrict__ wah) {
  int b = blockIdx.x;
  for (int a = threadIdx.x; a < kA; a += blockDim.x) {
    const float* w = wa_w + (size_t)a * kHD;
    const float* h = hidden + b * kHD;
    float s = wa_b[a];
    for (int i = 0; i < kHD; ++i) s += w[i] * h[i];
    wah[b * kA + a] = s;
  }
}

// ---------------- per step: energy = Va . tanh(Ua_fea + Wa_h + cover) + Va_b ----------------
// cover via im2col WMMA: A = 16 positions x 128 (121-tap window, bf16 in LDS, ds_load_b128),
// B = fused kernel (128 x 512 bf16, global b128). 8 waves x 4 N-tiles x 4 K-chunks.
__global__ void k_energy(const __bf16* __restrict__ ua_feaT, const __bf16* __restrict__ kf,
                         const float* __restrict__ alpha_sum, const float* __restrict__ wah,
                         const float* __restrict__ uf_b, const float* __restrict__ va_w,
                         const float* __restrict__ va_b, float* __restrict__ energy) {
  __shared__ __bf16 awin[16][128];
  __shared__ float sm_e[16];
  int tid = threadIdx.x;
  int mt = blockIdx.x;
  int p0 = mt * 16;
  int b = p0 >> 10;
  int hw0 = p0 & 1023;

  if (tid < 16) sm_e[tid] = 0.f;
  // gather 11x11 windows of alpha_sum into the A operand (zero-padded taps 121..127)
  for (int idx = tid; idx < 16 * 128; idx += 256) {
    int r = idx >> 7, c = idx & 127;
    float v = 0.f;
    if (c < 121) {
      int hw = hw0 + r;
      int h = hw >> 6, w = hw & 63;
      int i = c / 11, j = c % 11;
      int hh = h + i - 5, ww = w + j - 5;
      if (hh >= 0 && hh < kH && ww >= 0 && ww < kW) v = alpha_sum[b * kHW + hh * kW + ww];
    }
    awin[r][c] = (__bf16)v;
  }
  __syncthreads();

  int wave = tid >> 5, lane = tid & 31;
  int hi = lane >> 4, ln = lane & 15;
  const __bf16* awrow = &awin[ln][0];
  for (int nt = 0; nt < 4; ++nt) {
    int aCol = wave * 64 + nt * 16 + ln;
    const __bf16* krow = kf + (size_t)aCol * 128;
    __builtin_prefetch(ua_feaT + (size_t)aCol * kP + p0, 0, 1);   // global_prefetch_b8
    v8f acc = {};
#pragma unroll
    for (int kc = 0; kc < 4; ++kc) {
      v8bf a0 = *(const v8bf*)(awrow + kc * 32 + hi * 8);         // ds_load_b128
      v8bf a1 = *(const v8bf*)(awrow + kc * 32 + 16 + hi * 8);
      v16bf af = __builtin_shufflevector(a0, a1, 0, 1, 2, 3, 4, 5, 6, 7,
                                         8, 9, 10, 11, 12, 13, 14, 15);
      v8bf b0 = *(const v8bf*)(krow + kc * 32 + hi * 8);
      v8bf b1 = *(const v8bf*)(krow + kc * 32 + 16 + hi * 8);
      v16bf bfm = __builtin_shufflevector(b0, b1, 0, 1, 2, 3, 4, 5, 6, 7,
                                          8, 9, 10, 11, 12, 13, 14, 15);
      acc = __builtin_amdgcn_wmma_f32_16x16x32_bf16(false, af, false, bfm, (short)0, acc,
                                                    false, false);
    }
    float wv = wah[b * kA + aCol];
    float ub = uf_b[aCol];
    float vav = va_w[aCol];
    // Ua_fea for this column, rows m = hi*8..hi*8+7: contiguous -> one b128 load
    v8bf uf8 = *(const v8bf*)(ua_feaT + (size_t)aCol * kP + p0 + hi * 8);
#pragma unroll
    for (int v = 0; v < 8; ++v) {
      int m = v + hi * 8;
      float sc = tanhf((float)uf8[v] + wv + acc[v] + ub);
      atomicAdd(&sm_e[m], vav * sc);   // ds_add_f32
    }
  }
  __syncthreads();
  if (tid < 16) energy[b * kHW + hw0 + tid] = sm_e[tid] + va_b[0];
}

// ---------------- per step: softmax over HW, update alpha_sum, emit alpha ----------------
__global__ void k_softmax(const float* __restrict__ energy, float* __restrict__ alpha,
                          float* __restrict__ alpha_sum, float* __restrict__ out_alpha, int t) {
  __shared__ float red[256];
  int b = blockIdx.x, tid = threadIdx.x;
  const float* e = energy + b * kHW;
  float mx = -1e30f;
  for (int i = tid; i < kHW; i += 256) mx = fmaxf(mx, e[i]);
  red[tid] = mx;
  __syncthreads();
  for (int s = 128; s > 0; s >>= 1) {
    if (tid < s) red[tid] = fmaxf(red[tid], red[tid + s]);
    __syncthreads();
  }
  float gmx = red[0];
  __syncthreads();
  float sum = 0.f;
  for (int i = tid; i < kHW; i += 256) {
    float ev = expf(e[i] - gmx);
    alpha[b * kHW + i] = ev;
    sum += ev;
  }
  red[tid] = sum;
  __syncthreads();
  for (int s = 128; s > 0; s >>= 1) {
    if (tid < s) red[tid] += red[tid + s];
    __syncthreads();
  }
  float inv = 1.0f / red[0];
  for (int i = tid; i < kHW; i += 256) {
    float a = alpha[b * kHW + i] * inv;
    alpha[b * kHW + i] = a;
    alpha_sum[b * kHW + i] += a;
    out_alpha[((size_t)b * kT + t) * kHW + i] = a;
  }
}

// ---------------- per step: context[b,d] = sum_hw alpha * feat ----------------
__global__ void k_context(const float* __restrict__ feat, const float* __restrict__ alpha,
                          float* __restrict__ ctx) {
  __shared__ float sa[kHW];
  int b = blockIdx.x;
  for (int i = threadIdx.x; i < kHW; i += blockDim.x) sa[i] = alpha[b * kHW + i];
  __syncthreads();
  int d = blockIdx.y * blockDim.x + threadIdx.x;
  if (d < kD) {
    const float* f = feat + ((size_t)b * kD + d) * kHW;
    float s = 0.f;
    for (int i = 0; i < kHW; ++i) s += sa[i] * f[i];
    ctx[b * kDP + d] = s;
  }
}

// ---------------- per step: GRU cell + output head + vocab projection ----------------
__global__ void k_gru(const int* __restrict__ labels, const float* __restrict__ emb,
                      const float* __restrict__ ctx, const float* __restrict__ gwih,
                      const float* __restrict__ gwhh, const float* __restrict__ gbih,
                      const float* __restrict__ gbhh, const float* __restrict__ hw_w,
                      const float* __restrict__ hw_b, const float* __restrict__ ew_w,
                      const float* __restrict__ ew_b, const float* __restrict__ cw_w,
                      const float* __restrict__ cw_b, const float* __restrict__ w0_w,
                      const float* __restrict__ w0_b, float* __restrict__ hidden,
                      float* __restrict__ out_prob, int t) {
  __shared__ float se[kE], sc[kD], sh[kHD], sgx[3 * kHD], sgh[3 * kHD], shn[kHD], sout[kE];
  int b = blockIdx.x, tid = threadIdx.x;
  int id = (t == 0) ? 1 : labels[b * kT + (t - 1)];
  for (int i = tid; i < kE; i += 256) se[i] = emb[(size_t)id * kE + i];
  for (int i = tid; i < kD; i += 256) sc[i] = ctx[b * kDP + i];
  for (int i = tid; i < kHD; i += 256) sh[i] = hidden[b * kHD + i];
  __syncthreads();
  for (int g = tid; g < 3 * kHD; g += 256) {
    const float* wih = gwih + (size_t)g * (kE + kD);
    float s = gbih[g];
    for (int i = 0; i < kE; ++i) s += wih[i] * se[i];
    for (int i = 0; i < kD; ++i) s += wih[kE + i] * sc[i];
    sgx[g] = s;
    const float* whh = gwhh + (size_t)g * kHD;
    float s2 = gbhh[g];
    for (int i = 0; i < kHD; ++i) s2 += whh[i] * sh[i];
    sgh[g] = s2;
  }
  __syncthreads();
  if (tid < kHD) {
    float r = 1.f / (1.f + expf(-(sgx[tid] + sgh[tid])));
    float z = 1.f / (1.f + expf(-(sgx[kHD + tid] + sgh[kHD + tid])));
    float nn = tanhf(sgx[2 * kHD + tid] + r * sgh[2 * kHD + tid]);
    shn[tid] = (1.f - z) * nn + z * sh[tid];
  }
  __syncthreads();
  if (tid < kE) {
    float s = hw_b[tid] + ew_b[tid] + cw_b[tid];
    const float* w1 = hw_w + (size_t)tid * kHD;
    const float* w2 = ew_w + (size_t)tid * kE;
    const float* w3 = cw_w + (size_t)tid * kD;
    for (int i = 0; i < kHD; ++i) s += w1[i] * shn[i];
    for (int i = 0; i < kE; ++i) s += w2[i] * se[i];
    for (int i = 0; i < kD; ++i) s += w3[i] * sc[i];
    sout[tid] = s;
    hidden[b * kHD + tid] = shn[tid];
  }
  __syncthreads();
  for (int v = tid; v < kV; v += 256) {
    const float* w = w0_w + (size_t)v * kE;
    float s = w0_b[v];
    for (int i = 0; i < kE; ++i) s += w[i] * sout[i];
    out_prob[((size_t)b * kT + t) * kV + v] = s;
  }
}

extern "C" void kernel_launch(void* const* d_in, const int* in_sizes, int n_in,
                              void* d_out, int out_size, void* d_ws, size_t ws_size,
                              hipStream_t stream) {
  (void)in_sizes; (void)n_in; (void)out_size; (void)ws_size;
  const float* feat   = (const float*)d_in[0];
  const float* convq  = (const float*)d_in[2];
  const float* ua_w   = (const float*)d_in[3];
  const float* ua_b   = (const float*)d_in[4];
  const float* wa_w   = (const float*)d_in[5];
  const float* wa_b   = (const float*)d_in[6];
  const float* uf_w   = (const float*)d_in[7];
  const float* uf_b   = (const float*)d_in[8];
  const float* va_w   = (const float*)d_in[9];
  const float* va_b   = (const float*)d_in[10];
  const float* init_w = (const float*)d_in[11];
  const float* init_b = (const float*)d_in[12];
  const float* emb    = (const float*)d_in[13];
  const float* gwih   = (const float*)d_in[14];
  const float* gwhh   = (const float*)d_in[15];
  const float* gbih   = (const float*)d_in[16];
  const float* gbhh   = (const float*)d_in[17];
  const float* hw_w   = (const float*)d_in[18];
  const float* hw_b   = (const float*)d_in[19];
  const float* ew_w   = (const float*)d_in[20];
  const float* ew_b   = (const float*)d_in[21];
  const float* cw_w   = (const float*)d_in[22];
  const float* cw_b   = (const float*)d_in[23];
  const float* w0_w   = (const float*)d_in[24];
  const float* w0_b   = (const float*)d_in[25];
  const int*   labels = (const int*)d_in[26];

  char* ws = (char*)d_ws;
  size_t off = 0;
  auto alloc = [&](size_t bytes) -> void* {
    void* p = ws + off;
    off = (off + bytes + 255) & ~(size_t)255;
    return p;
  };
  __bf16* featT    = (__bf16*)alloc((size_t)kP * kDP * 2);   // 46 MB
  __bf16* uawb     = (__bf16*)alloc((size_t)kA * kDP * 2);
  __bf16* ua_feaT  = (__bf16*)alloc((size_t)kA * kP * 2);    // 32 MB, L2-resident
  __bf16* kf       = (__bf16*)alloc((size_t)kA * 128 * 2);
  float* alpha_sum = (float*)alloc((size_t)kB * kHW * 4);
  float* energy    = (float*)alloc((size_t)kB * kHW * 4);
  float* alpha     = (float*)alloc((size_t)kB * kHW * 4);
  float* hidden    = (float*)alloc((size_t)kB * kHD * 4);
  float* wah       = (float*)alloc((size_t)kB * kA * 4);
  float* ctx       = (float*)alloc((size_t)kB * kDP * 4);
  float* avg       = (float*)alloc((size_t)kB * kDP * 4);

  float* out_prob  = (float*)d_out;
  float* out_alpha = out_prob + (size_t)kB * kT * kV;

  hipMemsetAsync(alpha_sum, 0, (size_t)kB * kHW * 4, stream);

  k_avg<<<kB, 256, 0, stream>>>(feat, avg);
  k_hidden0<<<kB, 256, 0, stream>>>(avg, init_w, init_b, hidden);
  k_featT<<<dim3(kMT, kDP / 16), dim3(16, 16), 0, stream>>>(feat, featT);
  k_uawcvt<<<kA, 256, 0, stream>>>(ua_w, uawb);
  k_fusecover<<<kA, 128, 0, stream>>>(uf_w, convq, kf);
  k_uafea<<<dim3(kMT, kA / 16), 32, 0, stream>>>(featT, uawb, ua_b, ua_feaT);

  for (int t = 0; t < kT; ++t) {
    k_wah<<<kB, 256, 0, stream>>>(hidden, wa_w, wa_b, wah);
    k_energy<<<kMT, 256, 0, stream>>>(ua_feaT, kf, alpha_sum, wah, uf_b, va_w, va_b, energy);
    k_softmax<<<kB, 256, 0, stream>>>(energy, alpha, alpha_sum, out_alpha, t);
    k_context<<<dim3(kB, 6), 128, 0, stream>>>(feat, alpha, ctx);
    k_gru<<<kB, 256, 0, stream>>>(labels, emb, ctx, gwih, gwhh, gbih, gbhh, hw_w, hw_b,
                                  ew_w, ew_b, cw_w, cw_b, w0_w, w0_b, hidden, out_prob, t);
  }
}
